// SpikingAttention_53395033424079
// MI455X (gfx1250) — matrix operand order repaired and checked by reading, not compiled, not run on hardware
//
#include <hip/hip_runtime.h>

// ---------------------------------------------------------------------------
// Spiking attention, MI455X (gfx1250), wave32 + WMMA bf16 + async-to-LDS.
//   B=1, S=2048, D=2048, H=16, hd=128.
// Pipeline (all on `stream`, bf16 intermediates in workspace, ~64 MB):
//   1) x -> bf16                       (xb,  [S][D])
//   2) Wq/Wk/Wv/Wo -> bf16 transposed  (wT,  [N][K])
//   3) fused QKV GEMM (32x32/wave) + spike + RoPE
//   4) flash attention: async double-buffered K tiles in LDS, online softmax
//   5) ctx @ WoT (32x32/wave) -> fp32 d_out
// ---------------------------------------------------------------------------

typedef __bf16 bf16_t;
typedef __attribute__((ext_vector_type(16))) __bf16        v16bf;
typedef __attribute__((ext_vector_type(8)))  float         v8f;
typedef __attribute__((ext_vector_type(4)))  unsigned int  u32x4;

#define SEQ 2048
#define DIM 2048
#define NH  16
#define HD  128

// Build a 16-bf16 fragment from two 16-byte chunks (call sites pass p1 = p0 + 8, giving
// contiguous k = 16*hi + s in A and B alike; all call sites are 16B-aligned by construction).
__device__ __forceinline__ v16bf ldfrag(const bf16_t* p0, const bf16_t* p1) {
  union { v16bf v; u32x4 q[2]; } u;
  u.q[0] = *reinterpret_cast<const u32x4*>(p0);
  u.q[1] = *reinterpret_cast<const u32x4*>(p1);
  return u.v;
}

__device__ __forceinline__ v8f wmma_bf16(v16bf a, v16bf b, v8f c) {
  return __builtin_amdgcn_wmma_f32_16x16x32_bf16(
      /*neg_a=*/false, a, /*neg_b=*/false, b,
      /*c_mod=*/(short)0, c, /*reuse_a=*/false, /*reuse_b=*/false);
}

// Async copy of one 32-key x 128-ch bf16 K tile (8 KB) into this wave's LDS
// slot: 16 x global_load_async_to_lds_b128 (each lane 16 B => 512 B/instr).
// Tracked by ASYNCcnt; caller synchronizes with s_wait_asynccnt.
__device__ __forceinline__ void async_ktile(const bf16_t* __restrict__ rk,
                                            bf16_t* kslot, int kb, int h,
                                            int lane) {
  const char* gbase = (const char*)(rk + (size_t)kb * DIM + h * HD);
  char* lbase = (char*)kslot;
#pragma unroll
  for (int it = 0; it < 16; ++it) {
    int off  = lane * 16 + it * 512;  // flat byte offset in tile
    int row  = off >> 8;              // key row (256 B per row chunk)
    int colb = off & 255;             // byte within row
    unsigned long long ga =
        (unsigned long long)(uintptr_t)(gbase + (size_t)row * (DIM * 2) + colb);
    unsigned la = (unsigned)(uintptr_t)(lbase + row * 256 + colb);
    asm volatile("global_load_async_to_lds_b128 %0, %1, off"
                 :: "v"(la), "v"(ga) : "memory");
  }
}

// ---------------------------------------------------------------------------
// 1) fp32 -> bf16 elementwise
// ---------------------------------------------------------------------------
__global__ void k_cvt(const float* __restrict__ src, bf16_t* __restrict__ dst,
                      int n) {
  for (int i = blockIdx.x * blockDim.x + threadIdx.x; i < n;
       i += gridDim.x * blockDim.x)
    dst[i] = (bf16_t)src[i];
}

// ---------------------------------------------------------------------------
// 2) fp32 W[k][n] -> bf16 Wt[n][k]  (LDS-tiled 32x32 transpose)
// ---------------------------------------------------------------------------
__global__ void k_cvt_t(const float* __restrict__ w, bf16_t* __restrict__ wt) {
  __shared__ float tile[32][33];
  int bx = blockIdx.x * 32;  // n block
  int by = blockIdx.y * 32;  // k block
  int tx = threadIdx.x, ty = threadIdx.y;  // (32, 8)
  for (int r = ty; r < 32; r += 8)
    tile[r][tx] = w[(by + r) * DIM + bx + tx];
  __syncthreads();
  for (int r = ty; r < 32; r += 8)
    wt[(bx + r) * DIM + by + tx] = (bf16_t)tile[tx][r];
}

// ---------------------------------------------------------------------------
// 3) QKV GEMM + spike + RoPE.  One wave computes a 32x32 output region:
//    2 row tiles x (col tile c and its rotate-half partner c+64) => 4 WMMAs
//    per k-step sharing 2 A- and 2 B-fragments (2 x b128 chunks per WMMA).
//    12288 waves = 3 mats * 16 heads * 4 col-pairs * 64 row-pairs.
// ---------------------------------------------------------------------------
__global__ void k_qkv(const bf16_t* __restrict__ xb,
                      const bf16_t* __restrict__ wqT,
                      const bf16_t* __restrict__ wkT,
                      const bf16_t* __restrict__ wvT,
                      bf16_t* __restrict__ rq, bf16_t* __restrict__ rk,
                      bf16_t* __restrict__ svT) {
  int wid  = blockIdx.x * (blockDim.x >> 5) + (threadIdx.x >> 5);
  int lane = threadIdx.x & 31;
  int l15 = lane & 15, hi = lane >> 4;

  int rt2 = wid & 63; wid >>= 6;
  int cp  = wid & 3;  wid >>= 2;
  int h   = wid & 15;
  int mat = wid >> 4;  // 0=q, 1=k, 2=v

  const bf16_t* wt = (mat == 0) ? wqT : (mat == 1) ? wkT : wvT;

  int row0 = rt2 * 32;
  int col0 = h * HD + cp * 16;  // first-half column tile of this head
  int col1 = col0 + 64;         // rotate-half partner tile

  v8f a00 = {}, a01 = {}, a10 = {}, a11 = {};
  const bf16_t* ar0  = xb + (size_t)(row0 + l15) * DIM + 16 * hi;
  const bf16_t* ar1  = ar0 + 16 * DIM;
  const bf16_t* b0row = wt + (size_t)(col0 + l15) * DIM + 16 * hi;
  const bf16_t* b1row = wt + (size_t)(col1 + l15) * DIM + 16 * hi;
  for (int k0 = 0; k0 < DIM; k0 += 32) {
    v16bf A0 = ldfrag(ar0 + k0, ar0 + k0 + 8);
    v16bf A1 = ldfrag(ar1 + k0, ar1 + k0 + 8);
    v16bf B0 = ldfrag(b0row + k0, b0row + k0 + 8);
    v16bf B1 = ldfrag(b1row + k0, b1row + k0 + 8);
    a00 = wmma_bf16(A0, B0, a00);
    a01 = wmma_bf16(A0, B1, a01);
    a10 = wmma_bf16(A1, B0, a10);
    a11 = wmma_bf16(A1, B1, a11);
  }

  int dl = cp * 16 + l15;  // in-head dim, < 64
  // inv_freq[dl] = 10000^(-dl/64) = 2^(-dl * log2(10000)/64)
  float invf = exp2f(-(float)dl * 0.20762050593045702f);

#pragma unroll
  for (int half = 0; half < 2; ++half) {
    v8f acc0 = half ? a10 : a00;
    v8f acc1 = half ? a11 : a01;
    int rbase = row0 + 16 * half;
    if (mat == 2) {
      // spike(v), store transposed per head: svT[h][d][s] (16B packed stores)
      union { bf16_t b[8]; u32x4 q; } p0, p1;
#pragma unroll
      for (int r = 0; r < 8; ++r) {
        p0.b[r] = (bf16_t)((acc0[r] > 1.0f) ? 1.0f : 0.0f);
        p1.b[r] = (bf16_t)((acc1[r] > 1.0f) ? 1.0f : 0.0f);
      }
      *reinterpret_cast<u32x4*>(
          &svT[(size_t)(h * HD + dl) * SEQ + rbase + 8 * hi]) = p0.q;
      *reinterpret_cast<u32x4*>(
          &svT[(size_t)(h * HD + dl + 64) * SEQ + rbase + 8 * hi]) = p1.q;
    } else {
      bf16_t* dst = (mat == 0) ? rq : rk;
#pragma unroll
      for (int r = 0; r < 8; ++r) {
        int srow = rbase + r + 8 * hi;
        float s0 = (acc0[r] > 1.0f) ? 1.0f : 0.0f;
        float s1 = (acc1[r] > 1.0f) ? 1.0f : 0.0f;
        float sn, cs;
        __sincosf((float)srow * invf, &sn, &cs);
        dst[(size_t)srow * DIM + col0 + l15] = (bf16_t)(s0 * cs - s1 * sn);
        dst[(size_t)srow * DIM + col1 + l15] = (bf16_t)(s1 * cs + s0 * sn);
      }
    }
  }
}

// ---------------------------------------------------------------------------
// 4) Flash attention: one wave per (head, 16-query tile); 32-key blocks.
//    K tiles are double-buffer prefetched into LDS with
//    global_load_async_to_lds_b128 (ASYNCcnt), overlapping the softmax VALU
//    work of the current block; score B-fragments then come from LDS.
//    2 waves/block => 34 KB static LDS.
// ---------------------------------------------------------------------------
__global__ void k_attn(const bf16_t* __restrict__ rq,
                       const bf16_t* __restrict__ rk,
                       const bf16_t* __restrict__ svT,
                       bf16_t* __restrict__ ctx) {
  __shared__ __align__(16) bf16_t kstage[2][2][32 * 128];  // [wave][buf] 32 KB
  __shared__ __align__(16) bf16_t pbuf[2][16 * 32];        // 2 KB
  int wslot = threadIdx.x >> 5;  // 0..1
  int wid = blockIdx.x * 2 + wslot;
  int lane = threadIdx.x & 31;
  int l15 = lane & 15, hi = lane >> 4;

  int qt = wid & 127;
  int h  = wid >> 7;
  int q0 = qt * 16;

  // Query fragments: 16 x 128 as 4 A-fragments (16x32 bf16)
  v16bf aq[4];
#pragma unroll
  for (int j = 0; j < 4; ++j) {
    const bf16_t* ap = rq + (size_t)(q0 + l15) * DIM + h * HD + 32 * j + 16 * hi;
    aq[j] = ldfrag(ap, ap + 8);
  }

  const float kscale = 0.08838834764831843f * 1.4426950408889634f; // scale*log2e
  float mrow[8], lrow[8];
  v8f o[8];
  v8f zf = {};
#pragma unroll
  for (int r = 0; r < 8; ++r) { mrow[r] = -__builtin_inff(); lrow[r] = 0.0f; }
#pragma unroll
  for (int f = 0; f < 8; ++f) o[f] = zf;

  bf16_t* pb = pbuf[wslot];
  int nblk = (q0 + 16 + 31) >> 5;  // # of 32-key blocks

  async_ktile(rk, kstage[wslot][0], 0, h, lane);

  for (int b = 0; b < nblk; ++b) {
    int kb = b * 32;
    int cur = b & 1;
    bool more = (b + 1) < nblk;
    if (more) async_ktile(rk, kstage[wslot][1 - cur], kb + 32, h, lane);
    // current tile landed when <=16 (the next tile's ops) remain outstanding
    if (more) asm volatile("s_wait_asynccnt 0x10" ::: "memory");
    else      asm volatile("s_wait_asynccnt 0x0"  ::: "memory");

    const bf16_t* ks = kstage[wslot][cur];
    v8f s0 = {}, s1 = {};
#pragma unroll
    for (int j = 0; j < 4; ++j) {
      const bf16_t* kp0 = ks + l15 * 128 + 32 * j + 16 * hi;
      const bf16_t* kp1 = kp0 + 16 * 128;
      s0 = wmma_bf16(aq[j], ldfrag(kp0, kp0 + 8), s0);
      s1 = wmma_bf16(aq[j], ldfrag(kp1, kp1 + 8), s1);
    }
    // causal mask + online softmax (exp2 domain); row = r + 8*hi, col = l15
#pragma unroll
    for (int r = 0; r < 8; ++r) {
      int qi = q0 + r + 8 * hi;
      float t0 = (kb + l15 > qi)      ? -__builtin_inff() : s0[r] * kscale;
      float t1 = (kb + 16 + l15 > qi) ? -__builtin_inff() : s1[r] * kscale;
      float mx = fmaxf(t0, t1);
#pragma unroll
      for (int msk = 1; msk <= 8; msk <<= 1)
        mx = fmaxf(mx, __shfl_xor(mx, msk, 32));
      float mnew  = fmaxf(mrow[r], mx);
      float alpha = exp2f(mrow[r] - mnew);
      float p0 = exp2f(t0 - mnew);
      float p1 = exp2f(t1 - mnew);
      float ps = p0 + p1;
#pragma unroll
      for (int msk = 1; msk <= 8; msk <<= 1) ps += __shfl_xor(ps, msk, 32);
      lrow[r] = lrow[r] * alpha + ps;
      mrow[r] = mnew;
#pragma unroll
      for (int f = 0; f < 8; ++f) o[f][r] *= alpha;
      int ql = r + 8 * hi;
      pb[ql * 32 + l15]      = (bf16_t)p0;
      pb[ql * 32 + 16 + l15] = (bf16_t)p1;
    }
    // reload P in A-fragment layout (DS ops are in-order within a wave)
    v16bf pa = ldfrag(pb + l15 * 32 + 16 * hi, pb + l15 * 32 + 16 * hi + 8);
    // O += P @ V for 8 dim-tiles; svT gives contiguous B fragments
#pragma unroll
    for (int f = 0; f < 8; ++f) {
      const bf16_t* vp =
          svT + (size_t)(h * HD + 16 * f + l15) * SEQ + kb + 16 * hi;
      o[f] = wmma_bf16(pa, ldfrag(vp, vp + 8), o[f]);
    }
  }

#pragma unroll
  for (int r = 0; r < 8; ++r) {
    float invl = 1.0f / lrow[r];
    int srow = q0 + r + 8 * hi;
#pragma unroll
    for (int f = 0; f < 8; ++f)
      ctx[(size_t)srow * DIM + h * HD + 16 * f + l15] =
          (bf16_t)(o[f][r] * invl);
  }
}

// ---------------------------------------------------------------------------
// 5) out = ctx @ Wo (fp32 result); 32x32 region per wave (4 WMMAs/k-step)
// ---------------------------------------------------------------------------
__global__ void k_out(const bf16_t* __restrict__ ctx,
                      const bf16_t* __restrict__ woT,
                      float* __restrict__ out) {
  int wid  = blockIdx.x * (blockDim.x >> 5) + (threadIdx.x >> 5);
  int lane = threadIdx.x & 31;
  int l15 = lane & 15, hi = lane >> 4;
  int ct2 = wid & 63;
  int rt2 = wid >> 6;
  int row0 = rt2 * 32, col0 = ct2 * 32;

  v8f a00 = {}, a01 = {}, a10 = {}, a11 = {};
  const bf16_t* ar0 = ctx + (size_t)(row0 + l15) * DIM + 16 * hi;
  const bf16_t* ar1 = ar0 + 16 * DIM;
  const bf16_t* br0 = woT + (size_t)(col0 + l15) * DIM + 16 * hi;
  const bf16_t* br1 = br0 + 16 * DIM;
  for (int k0 = 0; k0 < DIM; k0 += 32) {
    v16bf A0 = ldfrag(ar0 + k0, ar0 + k0 + 8);
    v16bf A1 = ldfrag(ar1 + k0, ar1 + k0 + 8);
    v16bf B0 = ldfrag(br0 + k0, br0 + k0 + 8);
    v16bf B1 = ldfrag(br1 + k0, br1 + k0 + 8);
    a00 = wmma_bf16(A0, B0, a00);
    a01 = wmma_bf16(A0, B1, a01);
    a10 = wmma_bf16(A1, B0, a10);
    a11 = wmma_bf16(A1, B1, a11);
  }
#pragma unroll
  for (int r = 0; r < 8; ++r) {
    out[(size_t)(row0 + r + 8 * hi) * DIM + col0 + l15]           = a00[r];
    out[(size_t)(row0 + r + 8 * hi) * DIM + col0 + 16 + l15]      = a01[r];
    out[(size_t)(row0 + 16 + r + 8 * hi) * DIM + col0 + l15]      = a10[r];
    out[(size_t)(row0 + 16 + r + 8 * hi) * DIM + col0 + 16 + l15] = a11[r];
  }
}

// ---------------------------------------------------------------------------
extern "C" void kernel_launch(void* const* d_in, const int* in_sizes, int n_in,
                              void* d_out, int out_size, void* d_ws,
                              size_t ws_size, hipStream_t stream) {
  (void)in_sizes; (void)n_in; (void)out_size; (void)ws_size;

  const float* x  = (const float*)d_in[0];
  const float* Wq = (const float*)d_in[1];
  const float* Wk = (const float*)d_in[2];
  const float* Wv = (const float*)d_in[3];
  const float* Wo = (const float*)d_in[4];
  float* out = (float*)d_out;

  // Workspace layout (bf16, 8 MB per 2048x2048 array, ~64 MB total)
  bf16_t* ws  = (bf16_t*)d_ws;
  bf16_t* xb  = ws;                  // [S][D]   (re-used as ctx after QKV)
  bf16_t* wqT = xb  + SEQ * DIM;     // [N][K]
  bf16_t* wkT = wqT + DIM * DIM;
  bf16_t* wvT = wkT + DIM * DIM;
  bf16_t* woT = wvT + DIM * DIM;
  bf16_t* rq  = woT + DIM * DIM;     // [S][H*hd]
  bf16_t* rk  = rq  + SEQ * DIM;     // [S][H*hd]
  bf16_t* svT = rk  + SEQ * DIM;     // [H][hd][S]
  bf16_t* ctx = xb;                  // alias: x copy dead after k_qkv

  k_cvt<<<4096, 256, 0, stream>>>(x, xb, SEQ * DIM);
  dim3 tb(32, 8), tg(DIM / 32, DIM / 32);
  k_cvt_t<<<tg, tb, 0, stream>>>(Wq, wqT);
  k_cvt_t<<<tg, tb, 0, stream>>>(Wk, wkT);
  k_cvt_t<<<tg, tb, 0, stream>>>(Wv, wvT);
  k_cvt_t<<<tg, tb, 0, stream>>>(Wo, woT);

  k_qkv<<<3072, 128, 0, stream>>>(xb, wqT, wkT, wvT, rq, rk, svT);
  k_attn<<<1024, 64, 0, stream>>>(rq, rk, svT, ctx);
  k_out<<<512, 256, 0, stream>>>(ctx, woT, out);
}
